// MultiheadedSelfAttention_17609365914550
// MI455X (gfx1250) — matrix operand order repaired and checked by reading, not compiled, not run on hardware
//
#include <hip/hip_runtime.h>
#include <hip/hip_bf16.h>
#include <float.h>

// ---------------- types / helpers ----------------
typedef _Float16 h16;
typedef __attribute__((ext_vector_type(4)))  _Float16 v4h;
typedef __attribute__((ext_vector_type(8)))  _Float16 v8h;
typedef __attribute__((ext_vector_type(16))) _Float16 v16h;
typedef __attribute__((ext_vector_type(8)))  float    v8f;

#define BATCH 2
#define SEQ   2048
#define EMB   768
#define NH    12
#define HD    64
#define NTOK  (BATCH*SEQ)     /* 4096 */
#define FQKV  (3*EMB)         /* 2304 */
#define ATT_SCALE 0.125f      /* 64^-0.5 */

__device__ __forceinline__ v16h cat16(v8h a, v8h b) {
  return __builtin_shufflevector(a, b, 0,1,2,3,4,5,6,7,8,9,10,11,12,13,14,15);
}
__device__ __forceinline__ v8f wmma(v16h a, v16h b, v8f c) {
  // v_wmma_f32_16x16x32_f16: D = A(16x32 f16) * B(32x16 f16) + C(16x16 f32)
  return __builtin_amdgcn_wmma_f32_16x16x32_f16(false, a, false, b, (short)0, c,
                                                false, false);
}
__device__ __forceinline__ void split1(float f, h16& hi, h16& lo) {
  hi = (h16)f;
  lo = (h16)(f - (float)hi);
}
__device__ __forceinline__ void split4(const float4& v, v4h& hv, v4h& lv) {
  h16 a, b;
  split1(v.x, a, b); hv[0] = a; lv[0] = b;
  split1(v.y, a, b); hv[1] = a; lv[1] = b;
  split1(v.z, a, b); hv[2] = a; lv[2] = b;
  split1(v.w, a, b); hv[3] = a; lv[3] = b;
}

// ---- CDNA5 async global->LDS copy (16B per lane), tracked by ASYNCcnt ----
__device__ __forceinline__ void async_b128(h16* lds, const void* g) {
  uint32_t lds_off = (uint32_t)(uintptr_t)lds;   // low 32 bits = LDS byte offset
  asm volatile("global_load_async_to_lds_b128 %0, %1, off"
               :: "v"(lds_off), "v"(g) : "memory");
}
__device__ __forceinline__ void wait_async0() {
  asm volatile("s_wait_asynccnt 0x0" ::: "memory");
}

// ---------------- GEMM: C(T x N) = X(T x 768) @ W(N x 768)^T + bias ----------------
// Register-pipelined staging: next k-tile's global loads overlap the WMMAs.
// MODE 0: N=2304; split to hi/lo f16 Q/K/V (V transposed per 32-key tile).
// MODE 1: N=768; row-major fp32 out.
template<int MODE>
__global__ __launch_bounds__(256) void gemm_split_f16(
    const float* __restrict__ X, const float* __restrict__ W,
    const float* __restrict__ bias, float* __restrict__ Of32,
    h16* __restrict__ Qh_, h16* __restrict__ Ql_,
    h16* __restrict__ Kh_, h16* __restrict__ Kl_,
    h16* __restrict__ Vh_, h16* __restrict__ Vl_)
{
  // stride 40 halves (80B): 16B-aligned rows, conflict-free fragment reads
  __shared__ __align__(16) h16 Ah[64*40], Al[64*40];
  __shared__ __align__(16) h16 Bh[128*40], Bl[128*40];

  const int tid  = threadIdx.x;
  const int lane = tid & 31, wave = tid >> 5;
  const int wm = wave & 1, wn = wave >> 1;
  const int l15 = lane & 15, hi16 = (lane >> 4) & 1;
  const int m0 = blockIdx.x * 64, n0 = blockIdx.y * 128;

  v8f acc[2][2] = {};
  float4 ra[2], rb[4];

  auto load_tiles = [&](int kk) {
    for (int i = 0; i < 2; ++i) {
      int f4 = tid + 256*i;                 // 0..511
      int row = f4 >> 3, c4 = f4 & 7;
      ra[i] = *(const float4*)&X[(size_t)(m0+row)*EMB + kk + c4*4];
    }
    for (int i = 0; i < 4; ++i) {
      int f4 = tid + 256*i;                 // 0..1023
      int row = f4 >> 3, c4 = f4 & 7;
      rb[i] = *(const float4*)&W[(size_t)(n0+row)*EMB + kk + c4*4];
    }
  };
  load_tiles(0);

  for (int k0 = 0; k0 < EMB; k0 += 32) {
    // ---- convert staged registers into hi/lo f16 LDS tiles ----
    for (int i = 0; i < 2; ++i) {
      int f4 = tid + 256*i;
      int row = f4 >> 3, c4 = f4 & 7;
      v4h hv, lv; split4(ra[i], hv, lv);
      *(v4h*)&Ah[row*40 + c4*4] = hv;
      *(v4h*)&Al[row*40 + c4*4] = lv;
    }
    for (int i = 0; i < 4; ++i) {
      int f4 = tid + 256*i;
      int row = f4 >> 3, c4 = f4 & 7;
      v4h hv, lv; split4(rb[i], hv, lv);
      *(v4h*)&Bh[row*40 + c4*4] = hv;
      *(v4h*)&Bl[row*40 + c4*4] = lv;
    }
    __syncthreads();
    if (k0 + 32 < EMB) load_tiles(k0 + 32);   // overlap with frag loads + WMMA

    v16h ah[2], al[2], bhf[2], blf[2];
    for (int mt = 0; mt < 2; ++mt) {
      int r = wm*32 + mt*16 + l15;
      int off = hi16 * 8;                   // A: K = {0..7,16..23} + 8*hi16
      ah[mt] = cat16(*(const v8h*)&Ah[r*40 + off], *(const v8h*)&Ah[r*40 + 16 + off]);
      al[mt] = cat16(*(const v8h*)&Al[r*40 + off], *(const v8h*)&Al[r*40 + 16 + off]);
    }
    for (int nt = 0; nt < 2; ++nt) {
      int r = wn*32 + nt*16 + l15;
      int off = hi16 * 8;                   // B: K = {0..7,16..23} + 8*hi16
      bhf[nt] = cat16(*(const v8h*)&Bh[r*40 + off], *(const v8h*)&Bh[r*40 + off + 16]);
      blf[nt] = cat16(*(const v8h*)&Bl[r*40 + off], *(const v8h*)&Bl[r*40 + off + 16]);
    }
    for (int mt = 0; mt < 2; ++mt)
      for (int nt = 0; nt < 2; ++nt) {
        acc[mt][nt] = wmma(ah[mt], bhf[nt], acc[mt][nt]);  // hi*hi
        acc[mt][nt] = wmma(ah[mt], blf[nt], acc[mt][nt]);  // hi*lo
        acc[mt][nt] = wmma(al[mt], bhf[nt], acc[mt][nt]);  // lo*hi
      }
    __syncthreads();
  }

  for (int mt = 0; mt < 2; ++mt)
    for (int nt = 0; nt < 2; ++nt) {
      int n = n0 + wn*32 + nt*16 + l15;
      float bn = bias[n];
      for (int v = 0; v < 8; ++v) {
        int m = m0 + wm*32 + mt*16 + v + hi16*8;
        float val = acc[mt][nt][v] + bn;
        if (MODE == 0) {
          int sel = n / EMB, rem = n % EMB;
          int hh = rem >> 6, d = rem & 63;
          int b = m >> 11, s = m & (SEQ-1);
          size_t base = (size_t)(b*NH + hh)*SEQ*HD;
          h16 hv, lv; split1(val, hv, lv);
          if (sel == 0) {
            size_t i0 = base + (size_t)s*HD + d;
            Qh_[i0] = hv; Ql_[i0] = lv;
          } else if (sel == 1) {
            size_t i0 = base + (size_t)s*HD + d;
            Kh_[i0] = hv; Kl_[i0] = lv;
          } else {  // V transposed per 32-key tile: [kb][d][key]
            size_t i0 = base + (size_t)(s >> 5)*(HD*32) + (size_t)d*32 + (s & 31);
            Vh_[i0] = hv; Vl_[i0] = lv;
          }
        } else {
          Of32[(size_t)m*EMB + n] = val;
        }
      }
    }
}

// ---------------- flash attention ----------------
// 8 waves/block, 128 queries, 32-key steps. K/V tiles arrive via async
// global->LDS copies (ASYNCcnt), double-buffered to overlap with WMMA.
__global__ __launch_bounds__(256) void flash_attn(
    const h16* __restrict__ Qhi, const h16* __restrict__ Qlo,
    const h16* __restrict__ Khi, const h16* __restrict__ Klo,
    const h16* __restrict__ Vhi, const h16* __restrict__ Vlo,
    const int* __restrict__ mask, float* __restrict__ X2)
{
  __shared__ __align__(16) h16 Qh[128*72], Ql[128*72];     // Q tile, stride 72
  __shared__ __align__(16) h16 Kh[2][32*72], Kl[2][32*72]; // K tile [key][d], x2
  __shared__ __align__(16) h16 Vh[2][64*40], Vl[2][64*40]; // V tile [d][key], x2
  __shared__ __align__(16) h16 Ph[8*16*40], Pl[8*16*40];   // per-wave P relayout
  __shared__ float Msk[SEQ];                               // -FLT_MAX * mask[b,*]

  const int tid  = threadIdx.x;
  const int lane = tid & 31, wave = tid >> 5;
  const int l15 = lane & 15, hi16 = (lane >> 4) & 1;
  const int qx = blockIdx.x, h = blockIdx.y, b = blockIdx.z;
  const size_t headoff = (size_t)(b*NH + h) * SEQ * HD;
  const h16* Qhg = Qhi + headoff + (size_t)qx*128*HD;
  const h16* Qlg = Qlo + headoff + (size_t)qx*128*HD;
  const h16* Khg = Khi + headoff;
  const h16* Klg = Klo + headoff;
  const h16* Vhg = Vhi + headoff;   // + kb*HD*32 + d*32 + key
  const h16* Vlg = Vlo + headoff;

  // per-thread chunk coordinates for the staging copies
  const int kkey = tid >> 3, kch = (tid & 7) * 8;   // K: 32 rows x 8 chunks
  const int vd   = tid >> 2, vch = (tid & 3) * 8;   // V: 64 rows x 4 chunks

  // ---- stage Q (once) + first K/V tile, all async; mask bias -> LDS ----
  for (int i = 0; i < 4; ++i) {
    int c = tid + 256*i;                    // 0..1023 (row, chunk)
    int row = c >> 3, ch = (c & 7) * 8;
    async_b128(&Qh[row*72 + ch], Qhg + (size_t)row*HD + ch);
    async_b128(&Ql[row*72 + ch], Qlg + (size_t)row*HD + ch);
  }
  async_b128(&Kh[0][kkey*72 + kch], Khg + (size_t)kkey*HD + kch);
  async_b128(&Kl[0][kkey*72 + kch], Klg + (size_t)kkey*HD + kch);
  async_b128(&Vh[0][vd*40 + vch], Vhg + (size_t)vd*32 + vch);
  async_b128(&Vl[0][vd*40 + vch], Vlg + (size_t)vd*32 + vch);
  for (int i = 0; i < SEQ/256; ++i) {       // 8 ints per thread
    int k = tid + 256*i;
    Msk[k] = -FLT_MAX * (float)mask[b*SEQ + k];
  }
  wait_async0();
  __syncthreads();

  // per-wave Q A-fragments: 16 queries x 64 d = two K=32 slabs, hi/lo
  v16h qhf[2], qlf[2];
  for (int ds = 0; ds < 2; ++ds) {
    int r = wave*16 + l15;
    int off = ds*32 + hi16*8;
    qhf[ds] = cat16(*(const v8h*)&Qh[r*72 + off], *(const v8h*)&Qh[r*72 + off + 16]);
    qlf[ds] = cat16(*(const v8h*)&Ql[r*72 + off], *(const v8h*)&Ql[r*72 + off + 16]);
  }

  v8f o[4] = {};
  float mrow[8], lrow[8];
  for (int v = 0; v < 8; ++v) { mrow[v] = -FLT_MAX; lrow[v] = 0.f; }
  h16* pw  = Ph + wave*16*40;
  h16* plw = Pl + wave*16*40;

  for (int kb = 0; kb < SEQ/32; ++kb) {
    const int k0 = kb*32;
    const int cur = kb & 1;
    // ---- issue async copy of NEXT tile into the other buffer ----
    if (kb + 1 < SEQ/32) {
      const int nxt = cur ^ 1;
      size_t koff = (size_t)(kb+1)*32*HD;
      async_b128(&Kh[nxt][kkey*72 + kch], Khg + koff + (size_t)kkey*HD + kch);
      async_b128(&Kl[nxt][kkey*72 + kch], Klg + koff + (size_t)kkey*HD + kch);
      size_t voff = (size_t)(kb+1)*HD*32;
      async_b128(&Vh[nxt][vd*40 + vch], Vhg + voff + (size_t)vd*32 + vch);
      async_b128(&Vl[nxt][vd*40 + vch], Vlg + voff + (size_t)vd*32 + vch);
    }

    // mask bias per key column, from LDS (reference: qk += mask_k * -FLT_MAX)
    float bias0 = Msk[k0 + l15];
    float bias1 = Msk[k0 + 16 + l15];

    // ---- scores S = Q K^T over two 16-key N-tiles ----
    v8f sa[2] = {};
    for (int kt = 0; kt < 2; ++kt) {
      int key = kt*16 + l15;
      for (int ds = 0; ds < 2; ++ds) {
        int off = ds*32 + hi16*8;
        v16h kbh = cat16(*(const v8h*)&Kh[cur][key*72 + off],
                         *(const v8h*)&Kh[cur][key*72 + off + 16]);
        v16h kbl = cat16(*(const v8h*)&Kl[cur][key*72 + off],
                         *(const v8h*)&Kl[cur][key*72 + off + 16]);
        sa[kt] = wmma(qhf[ds], kbh, sa[kt]);
        sa[kt] = wmma(qhf[ds], kbl, sa[kt]);
        sa[kt] = wmma(qlf[ds], kbh, sa[kt]);
      }
    }

    // ---- online softmax (rows 0-7 in lanes 0-15, rows 8-15 in lanes 16-31) ----
    float s0[8], s1[8], tm[8], rs[8], alpha[8];
    for (int v = 0; v < 8; ++v) {
      s0[v] = sa[0][v]*ATT_SCALE + bias0;
      s1[v] = sa[1][v]*ATT_SCALE + bias1;
      tm[v] = fmaxf(s0[v], s1[v]);
    }
    for (int off = 1; off < 16; off <<= 1)
      for (int v = 0; v < 8; ++v)
        tm[v] = fmaxf(tm[v], __shfl_xor(tm[v], off, 16));
    for (int v = 0; v < 8; ++v) {
      float mn = fmaxf(mrow[v], tm[v]);
      alpha[v] = __expf(mrow[v] - mn);
      mrow[v]  = mn;
      s0[v] = __expf(s0[v] - mn);
      s1[v] = __expf(s1[v] - mn);
      rs[v] = s0[v] + s1[v];
    }
    for (int off = 1; off < 16; off <<= 1)
      for (int v = 0; v < 8; ++v)
        rs[v] += __shfl_xor(rs[v], off, 16);
    for (int v = 0; v < 8; ++v) lrow[v] = lrow[v]*alpha[v] + rs[v];
    for (int dt = 0; dt < 4; ++dt)
      for (int v = 0; v < 8; ++v) o[dt][v] *= alpha[v];

    // ---- P: C layout -> LDS -> A layout (intra-wave; LDS in-order per wave) ----
    for (int v = 0; v < 8; ++v) {
      int m = v + hi16*8;
      h16 a, bb;
      split1(s0[v], a, bb); pw[m*40 + l15]      = a; plw[m*40 + l15]      = bb;
      split1(s1[v], a, bb); pw[m*40 + 16 + l15] = a; plw[m*40 + 16 + l15] = bb;
    }
    asm volatile("" ::: "memory");
    int off8 = hi16*8;
    v16h pfh = cat16(*(const v8h*)&pw[l15*40 + off8],  *(const v8h*)&pw[l15*40 + 16 + off8]);
    v16h pfl = cat16(*(const v8h*)&plw[l15*40 + off8], *(const v8h*)&plw[l15*40 + 16 + off8]);

    // ---- O += P @ V ----
    for (int dt = 0; dt < 4; ++dt) {
      int d = dt*16 + l15;
      int koff = hi16*8;
      v16h vfh = cat16(*(const v8h*)&Vh[cur][d*40 + koff],
                       *(const v8h*)&Vh[cur][d*40 + koff + 16]);
      v16h vfl = cat16(*(const v8h*)&Vl[cur][d*40 + koff],
                       *(const v8h*)&Vl[cur][d*40 + koff + 16]);
      o[dt] = wmma(pfh, vfh, o[dt]);
      o[dt] = wmma(pfh, vfl, o[dt]);
      o[dt] = wmma(pfl, vfh, o[dt]);
    }

    // next tile's async writes must land; then all waves sync
    wait_async0();
    __syncthreads();
  }

  // ---- epilogue: O / l -> X2 in (B,S,E) layout (e = h*64 + d) ----
  float inv[8];
  for (int v = 0; v < 8; ++v) inv[v] = 1.f / lrow[v];
  for (int dt = 0; dt < 4; ++dt)
    for (int v = 0; v < 8; ++v) {
      int q = qx*128 + wave*16 + v + hi16*8;
      int d = dt*16 + l15;
      X2[(size_t)(b*SEQ + q)*EMB + h*HD + d] = o[dt][v] * inv[v];
    }
}

// ---------------- launch ----------------
extern "C" void kernel_launch(void* const* d_in, const int* in_sizes, int n_in,
                              void* d_out, int out_size, void* d_ws, size_t ws_size,
                              hipStream_t stream) {
  const float* X      = (const float*)d_in[0];
  const int*   mask   = (const int*)  d_in[1];
  const float* w_qkv  = (const float*)d_in[2];
  const float* b_qkv  = (const float*)d_in[3];
  const float* w_proj = (const float*)d_in[4];
  const float* b_proj = (const float*)d_in[5];
  float* out = (float*)d_out;

  const size_t HS = (size_t)BATCH*NH*SEQ*HD;   // 3,145,728 elements
  h16* Qhi = (h16*)d_ws;
  h16* Qlo = Qhi + HS;
  h16* Khi = Qlo + HS;
  h16* Klo = Khi + HS;
  h16* Vhi = Klo + HS;
  h16* Vlo = Vhi + HS;
  float* X2 = (float*)(Vlo + HS);              // ~37.7MB f16 + 12.6MB f32

  // 1) QKV projection + bias -> hi/lo f16 Q/K/V (V tile-transposed)
  gemm_split_f16<0><<<dim3(NTOK/64, FQKV/128), 256, 0, stream>>>(
      X, w_qkv, b_qkv, nullptr, Qhi, Qlo, Khi, Klo, Vhi, Vlo);
  // 2) flash attention with async double-buffered K/V staging
  flash_attn<<<dim3(SEQ/128, NH, BATCH), 256, 0, stream>>>(
      Qhi, Qlo, Khi, Klo, Vhi, Vlo, mask, X2);
  // 3) output projection + bias
  gemm_split_f16<1><<<dim3(NTOK/64, EMB/128), 256, 0, stream>>>(
      X2, w_proj, b_proj, out, nullptr, nullptr, nullptr, nullptr, nullptr, nullptr);
}
